// FeaturesMemory_22119081575294
// MI455X (gfx1250) — matrix operand order repaired and checked by reading, not compiled, not run on hardware
//
#include <hip/hip_runtime.h>
#include <hip/hip_bf16.h>

typedef __attribute__((ext_vector_type(16))) _Float16 v16h;
typedef __attribute__((ext_vector_type(8)))  _Float16 v8h;
typedef __attribute__((ext_vector_type(8)))  float    v8f;

#define BN_RS 0.9999950000374997f   // 1/sqrt(1 + 1e-5)

__device__ __forceinline__ v8f zero8f(){
  v8f z;
#pragma unroll
  for (int i=0;i<8;i++) z[i]=0.0f;
  return z;
}
__device__ __forceinline__ v16h zero16h(){
  v16h z;
#pragma unroll
  for (int i=0;i<16;i++) z[i]=(_Float16)0.0f;
  return z;
}

// A fragment: 16x32 f16, A row-major MxK (lda in elements).
// lane L: row = m0 + (L&15); halves 0..7 -> K = k0 + (L<16?0:8)+i ; halves 8..15 -> K = k0+16+(L<16?0:8)+i
__device__ __forceinline__ v16h load_a16(const _Float16* __restrict__ A, int lda, int m0, int k0, int lane){
  const int row = m0 + (lane & 15);
  const int hi  = (lane >> 4) & 1;
  const _Float16* p = A + (size_t)row * lda + k0 + hi * 8;
  v8h lo = *(const v8h*)(p);
  v8h h2 = *(const v8h*)(p + 16);
  v16h r;
#pragma unroll
  for (int i=0;i<8;i++){ r[i]=lo[i]; r[8+i]=h2[i]; }
  return r;
}

// B fragment: 32x16 f16. B supplied as N x K row-major (weight layout), ldb in elements.
// lane L: col = n0 + (L&15); halves 0..7 -> K = k0 + (L<16?0:8)+i ; halves 8..15 -> K = k0+16+(L<16?0:8)+i
__device__ __forceinline__ v16h load_bw16(const _Float16* __restrict__ B, int ldb, int n0, int k0, int lane){
  const int col = n0 + (lane & 15);
  const int hi  = (lane >> 4) & 1;
  const _Float16* p = B + (size_t)col * ldb + k0 + hi * 8;
  v8h lo = *(const v8h*)(p);
  v8h h2 = *(const v8h*)(p + 16);
  v16h r;
#pragma unroll
  for (int i=0;i<8;i++){ r[i]=lo[i]; r[8+i]=h2[i]; }
  return r;
}

// -------------------------------------------------------------------------
// Generic WMMA GEMM:  C[m,n] = act( (sum_k A[m,k]*Bw[n,k]) * cscale , gamma, beta )
// block = 256 threads = 8 waves arranged 4(M) x 2(N). Block tile 128 x 128.
// Wave tile 32(M) x 64(N): 2 A-frags x 4 B-frags -> 8 WMMAs per 32-wide k-step.
// K unrolled by 64 with ping-pong register double buffering (requires K%64==0).
// grid = (N/128, M/128). Output f16, ldc elements.
// -------------------------------------------------------------------------
__global__ void __launch_bounds__(256)
gemm_wmma(const _Float16* __restrict__ A, int lda,
          const _Float16* __restrict__ Bw, int ldb,
          _Float16* __restrict__ C, int ldc,
          int K, float cscale,
          const float* __restrict__ gamma,
          const float* __restrict__ beta,
          int relu)
{
  const int lane = threadIdx.x & 31;
  const int w    = threadIdx.x >> 5;
  const int m0   = blockIdx.y * 128 + (w & 3) * 32;
  const int n0   = blockIdx.x * 128 + (w >> 2) * 64;

  v8f acc[2][4];
#pragma unroll
  for (int i=0;i<2;i++)
#pragma unroll
    for (int j=0;j<4;j++) acc[i][j] = zero8f();

  v16h a0[2], b0[4], a1[2], b1[4];

  auto loadAB = [&](v16h (&av)[2], v16h (&bv)[4], int k){
#pragma unroll
    for (int i=0;i<2;i++) av[i] = load_a16(A, lda, m0 + i*16, k, lane);
#pragma unroll
    for (int j=0;j<4;j++) bv[j] = load_bw16(Bw, ldb, n0 + j*16, k, lane);
  };
  auto compute = [&](v16h (&av)[2], v16h (&bv)[4]){
#pragma unroll
    for (int i=0;i<2;i++)
#pragma unroll
      for (int j=0;j<4;j++)
        acc[i][j] = __builtin_amdgcn_wmma_f32_16x16x32_f16(
                      false, av[i], false, bv[j], (short)0, acc[i][j], false, false);
  };

  loadAB(a0, b0, 0);
  for (int k0 = 0; k0 < K; k0 += 64){
    // prefetch the streams 8 k-steps ahead (lowers to global_prefetch_b8)
    if (k0 + 256 < K){
      __builtin_prefetch((const void*)(A  + (size_t)(m0 + (lane & 15)) * lda + k0 + 256), 0, 1);
      __builtin_prefetch((const void*)(Bw + (size_t)(n0 + (lane & 15)) * ldb + k0 + 256), 0, 1);
    }
    loadAB(a1, b1, k0 + 32);          // issue next half-step's loads first
    compute(a0, b0);                  // WMMAs overlap outstanding loads
    if (k0 + 64 < K) loadAB(a0, b0, k0 + 64);
    compute(a1, b1);
  }

  const int hi = (lane >> 4) & 1;
#pragma unroll
  for (int i=0;i<2;i++){
#pragma unroll
    for (int j=0;j<4;j++){
      const int n = n0 + j*16 + (lane & 15);
      float s = 1.0f, o = 0.0f;
      if (gamma){ s = gamma[n] * BN_RS; o = beta[n]; }
#pragma unroll
      for (int r=0;r<8;r++){
        const int m = m0 + i*16 + r + hi*8;
        float x = acc[i][j][r] * cscale;
        x = x * s + o;
        if (relu) x = fmaxf(x, 0.0f);
        C[(size_t)m * ldc + n] = (_Float16)x;
      }
    }
  }
}

// -------------------------------------------------------------------------
// 3x3 SAME conv via implicit GEMM. Input: xcat (16384 x 1536 f16, pixel-major).
// Weights Wt: 768 x 13824 f16, row-major (o, tap*1536 + c). Output NCHW f32
// with BN scale/bias + ReLU.
// block = 256 (8 waves = 2 px-tiles x 4 o-tiles): 64 pixels (one image row)
// x 256 out-channels per block. Wave tile 32 px x 64 o: 2 A-frags x 4 B-frags
// -> 8 WMMAs per 32-wide k-step (1.5 b128 loads per WMMA).
// Loop: tap(9) x c(1536), border predicate hoisted per tap; c unrolled by 64
// with ping-pong double buffering. grid = (768/256, B*64).
// -------------------------------------------------------------------------
__global__ void __launch_bounds__(256)
conv3x3_wmma(const _Float16* __restrict__ xcat,
             const _Float16* __restrict__ Wt,
             const float* __restrict__ bg, const float* __restrict__ bb,
             float* __restrict__ out)
{
  const int lane = threadIdx.x & 31;
  const int w    = threadIdx.x >> 5;
  const int row  = blockIdx.y;            // b*64 + y
  const int b    = row >> 6;
  const int y    = row & 63;
  const int x0   = (w & 1) * 32;          // 32-px tile within the 64-px row
  const int o0   = blockIdx.x * 256 + (w >> 1) * 64;
  const int lr   = lane & 15;
  const int hi   = (lane >> 4) & 1;

  v8f acc[2][4];
#pragma unroll
  for (int i=0;i<2;i++)
#pragma unroll
    for (int j=0;j<4;j++) acc[i][j] = zero8f();

  v16h a0[2], a1[2], b0[4], b1[4];

  auto loadB4 = [&](v16h (&bv)[4], int k){
#pragma unroll
    for (int j=0;j<4;j++) bv[j] = load_bw16(Wt, 13824, o0 + j*16, k, lane);
  };
  auto loadA1 = [&](v16h &av, const _Float16* abase, int c, bool valid){
    if (valid){
      v8h lo = *(const v8h*)(abase + c);
      v8h h2 = *(const v8h*)(abase + c + 16);
#pragma unroll
      for (int i=0;i<8;i++){ av[i]=lo[i]; av[8+i]=h2[i]; }
    } else {
      av = zero16h();
    }
  };
  auto compute = [&](v16h (&av)[2], v16h (&bv)[4]){
#pragma unroll
    for (int i=0;i<2;i++)
#pragma unroll
      for (int j=0;j<4;j++)
        acc[i][j] = __builtin_amdgcn_wmma_f32_16x16x32_f16(
                      false, av[i], false, bv[j], (short)0, acc[i][j], false, false);
  };

  for (int tap = 0; tap < 9; tap++){
    const int dy = tap / 3 - 1;
    const int dx = tap - (tap / 3) * 3 - 1;
    const int yy = y + dy;
    // two 16-px sub-tiles, each with its own border predicate / base pointer
    const int xxA = x0 + lr + dx;
    const int xxB = x0 + 16 + lr + dx;
    const bool vA = ((unsigned)yy < 64u) && ((unsigned)xxA < 64u);
    const bool vB = ((unsigned)yy < 64u) && ((unsigned)xxB < 64u);
    const _Float16* abA = xcat + ((size_t)((b*64 + yy)*64 + xxA)) * 1536 + hi * 8;
    const _Float16* abB = xcat + ((size_t)((b*64 + yy)*64 + xxB)) * 1536 + hi * 8;
    const int kb = tap * 1536;

    loadA1(a0[0], abA, 0, vA);
    loadA1(a0[1], abB, 0, vB);
    loadB4(b0, kb);
    for (int c0 = 0; c0 < 1536; c0 += 64){
      // prefetch the weight + activation streams ahead (global_prefetch_b8)
      if (c0 + 256 < 1536){
        __builtin_prefetch((const void*)(Wt + (size_t)(o0 + lr) * 13824 + kb + c0 + 256), 0, 1);
        if (vA) __builtin_prefetch((const void*)(abA + c0 + 256), 0, 1);
      }
      loadA1(a1[0], abA, c0 + 32, vA);
      loadA1(a1[1], abB, c0 + 32, vB);
      loadB4(b1, kb + c0 + 32);
      compute(a0, b0);
      if (c0 + 64 < 1536){
        loadA1(a0[0], abA, c0 + 64, vA);
        loadA1(a0[1], abB, c0 + 64, vB);
        loadB4(b0, kb + c0 + 64);
      }
      compute(a1, b1);
    }
  }

  const float rs = BN_RS;
#pragma unroll
  for (int i=0;i<2;i++){
#pragma unroll
    for (int j=0;j<4;j++){
      const int o = o0 + j*16 + lr;
      const float s  = bg[o] * rs;
      const float of = bb[o];
#pragma unroll
      for (int r=0;r<8;r++){
        const int px = x0 + i*16 + r + hi*8;
        const float v = fmaxf(acc[i][j][r] * s + of, 0.0f);
        out[(((size_t)b*768 + o) * 4096) + y*64 + px] = v;
      }
    }
  }
}

// ------------------------- small support kernels --------------------------

__global__ void cvt_f16(const float* __restrict__ s, _Float16* __restrict__ d, int n){
  int i = blockIdx.x * blockDim.x + threadIdx.x;
  if (i < n) d[i] = (_Float16)s[i];
}

// bw (768,1536,3,3) f32 -> Wt (768 x 13824) f16, k = tap*1536 + c
__global__ void cvt_conv_w(const float* __restrict__ s, _Float16* __restrict__ d){
  int i = blockIdx.x * blockDim.x + threadIdx.x;
  const int total = 768 * 13824;
  if (i < total){
    const int o   = i / 13824;
    const int k   = i - o * 13824;
    const int tap = k / 1536;
    const int c   = k - tap * 1536;
    d[i] = (_Float16) s[((size_t)(o * 1536 + c)) * 9 + tap];
  }
}

// feats NCHW f32 -> xcat[:, 0:768] f16 (pixel-major, ldc 1536)
__global__ void feats_to_xcat(const float* __restrict__ f, _Float16* __restrict__ xcat){
  int i = blockIdx.x * blockDim.x + threadIdx.x;
  const int total = 4 * 768 * 4096;
  if (i < total){
    const int b   = i / (768*4096);
    const int rem = i - b * 768*4096;
    const int c   = rem / 4096;
    const int hw  = rem - c * 4096;
    xcat[((size_t)(b*4096 + hw)) * 1536 + c] = (_Float16)f[i];
  }
}

// vh (P x 256, pixel-major) -> vT ((b*256+c) x 4096)
__global__ void v_transpose(const _Float16* __restrict__ vh, _Float16* __restrict__ vT){
  int i = blockIdx.x * blockDim.x + threadIdx.x;
  const int total = 4 * 256 * 4096;
  if (i < total){
    const int b   = i / (256*4096);
    const int rem = i - b * 256*4096;
    const int c   = rem / 4096;
    const int j   = rem - c * 4096;
    vT[i] = vh[((size_t)(b*4096 + j)) * 256 + c];
  }
}

// cosine-sim argmax over text_mem, gather memory rows into DhT (768 x 256 f16, zero-padded)
__global__ void sim_gather(const float* __restrict__ text,
                           const float* __restrict__ text_mem,
                           const float* __restrict__ memry,
                           _Float16* __restrict__ DhT)
{
  __shared__ float sv[256];
  __shared__ int   si[256];
  __shared__ int   s_best;
  const int i = blockIdx.x;     // 0..255
  const int t = threadIdx.x;    // 256
  if (i < 200){
    float best = -1e30f; int bidx = 0;
    if (t < 171){
      float dot = 0.f, ni = 0.f, nj = 0.f;
      for (int c=0;c<768;c++){
        const float a = text[i*768 + c];
        const float b = text_mem[t*768 + c];
        dot += a*b; ni += a*a; nj += b*b;
      }
      const float den = sqrtf(ni) * sqrtf(nj);
      best = dot / fmaxf(den, 1e-8f);
      bidx = t;
    }
    sv[t] = best; si[t] = bidx; __syncthreads();
    for (int s=128; s>0; s>>=1){
      if (t < s){
        if (sv[t+s] > sv[t]){ sv[t]=sv[t+s]; si[t]=si[t+s]; }
      }
      __syncthreads();
    }
    if (t == 0) s_best = si[0];
    __syncthreads();
    const int idx = s_best;
    for (int c=t; c<768; c+=256)
      DhT[(size_t)c*256 + i] = (_Float16)memry[idx*768 + c];
  } else {
    for (int c=t; c<768; c+=256)
      DhT[(size_t)c*256 + i] = (_Float16)0.0f;
  }
}

// per-pixel softmax over 200 classes -> wcls (P x 256 f16, zero-padded)
__global__ void softmax_wcls(const float* __restrict__ preds, _Float16* __restrict__ wcls){
  __shared__ float red[64];
  const int p = blockIdx.x;      // 16384
  const int t = threadIdx.x;     // 64
  const int b = p >> 12, hw = p & 4095;
  const float* src = preds + ((size_t)b * 200) * 4096 + hw;
  float v[4];
  float mx = -1e30f;
#pragma unroll
  for (int u=0;u<4;u++){
    const int n = t + u*64;
    v[u] = (n < 200) ? src[(size_t)n * 4096] : -1e30f;
    mx = fmaxf(mx, v[u]);
  }
  red[t] = mx; __syncthreads();
  for (int s=32;s>0;s>>=1){ if (t<s) red[t]=fmaxf(red[t], red[t+s]); __syncthreads(); }
  mx = red[0]; __syncthreads();
  float sum = 0.f;
#pragma unroll
  for (int u=0;u<4;u++){
    const int n = t + u*64;
    v[u] = (n < 200) ? __expf(v[u]-mx) : 0.f;
    sum += v[u];
  }
  red[t] = sum; __syncthreads();
  for (int s=32;s>0;s>>=1){ if (t<s) red[t]+=red[t+s]; __syncthreads(); }
  const float inv = 1.f / red[0];
#pragma unroll
  for (int u=0;u<4;u++){
    const int n = t + u*64;                  // covers 0..255
    wcls[(size_t)p*256 + n] = (_Float16)(v[u]*inv);   // v[u]==0 for n>=200
  }
}

// in-place row softmax over 4096-wide f16 rows (grid = 4096 rows)
__global__ void softmax_att(_Float16* __restrict__ att){
  __shared__ float red[256];
  _Float16* row = att + (size_t)blockIdx.x * 4096;
  const int t = threadIdx.x;
  float mx = -1e30f;
  for (int j=t; j<4096; j+=256) mx = fmaxf(mx, (float)row[j]);
  red[t] = mx; __syncthreads();
  for (int s=128;s>0;s>>=1){ if (t<s) red[t]=fmaxf(red[t], red[t+s]); __syncthreads(); }
  mx = red[0]; __syncthreads();
  float sum = 0.f;
  for (int j=t; j<4096; j+=256) sum += __expf((float)row[j]-mx);
  red[t] = sum; __syncthreads();
  for (int s=128;s>0;s>>=1){ if (t<s) red[t]+=red[t+s]; __syncthreads(); }
  const float inv = 1.f / red[0];
  for (int j=t; j<4096; j+=256)
    row[j] = (_Float16)(__expf((float)row[j]-mx) * inv);
}

// -------------------------------------------------------------------------

extern "C" void kernel_launch(void* const* d_in, const int* in_sizes, int n_in,
                              void* d_out, int out_size, void* d_ws, size_t ws_size,
                              hipStream_t stream)
{
  const float* feats    = (const float*)d_in[0];
  const float* preds    = (const float*)d_in[1];
  const float* text     = (const float*)d_in[2];
  const float* memry    = (const float*)d_in[3];
  const float* text_mem = (const float*)d_in[4];
  const float* qw1 = (const float*)d_in[5];  const float* qg1 = (const float*)d_in[6];  const float* qb1 = (const float*)d_in[7];
  const float* qw2 = (const float*)d_in[8];  const float* qg2 = (const float*)d_in[9];  const float* qb2 = (const float*)d_in[10];
  const float* kw1 = (const float*)d_in[11]; const float* kg1 = (const float*)d_in[12]; const float* kb1 = (const float*)d_in[13];
  const float* kw2 = (const float*)d_in[14]; const float* kg2 = (const float*)d_in[15]; const float* kb2 = (const float*)d_in[16];
  const float* vw  = (const float*)d_in[17]; const float* vg  = (const float*)d_in[18]; const float* vb  = (const float*)d_in[19];
  const float* ow  = (const float*)d_in[20]; const float* og  = (const float*)d_in[21]; const float* ob  = (const float*)d_in[22];
  const float* bw  = (const float*)d_in[23]; const float* bg  = (const float*)d_in[24]; const float* bb  = (const float*)d_in[25];
  float* out = (float*)d_out;

  // ---- workspace layout ----
  char* base = (char*)d_ws;
  size_t off = 0;
  auto alloc = [&](size_t bytes) -> void* {
    off = (off + 255) & ~(size_t)255;
    void* p = base + off;
    off += bytes;
    return p;
  };
  const size_t P = 16384;
  _Float16* DhT   = (_Float16*)alloc((size_t)768*256*2);
  _Float16* wcls  = (_Float16*)alloc(P*256*2);
  _Float16* xcat  = (_Float16*)alloc(P*1536*2);
  _Float16* sel   = (_Float16*)alloc(P*768*2);
  _Float16* q1b   = (_Float16*)alloc(P*256*2);
  _Float16* k1b   = (_Float16*)alloc(P*256*2);
  _Float16* qh    = (_Float16*)alloc(P*256*2);
  _Float16* kh    = (_Float16*)alloc(P*256*2);
  _Float16* vh    = (_Float16*)alloc(P*256*2);
  _Float16* vT    = (_Float16*)alloc((size_t)4*256*4096*2);
  _Float16* atth  = (_Float16*)alloc((size_t)4096*4096*2);
  _Float16* ctxh  = (_Float16*)alloc(P*256*2);
  _Float16* qw1h  = (_Float16*)alloc((size_t)256*768*2);
  _Float16* qw2h  = (_Float16*)alloc((size_t)256*256*2);
  _Float16* kw1h  = (_Float16*)alloc((size_t)256*768*2);
  _Float16* kw2h  = (_Float16*)alloc((size_t)256*256*2);
  _Float16* vwh   = (_Float16*)alloc((size_t)256*768*2);
  _Float16* owh   = (_Float16*)alloc((size_t)768*256*2);
  _Float16* bwh   = (_Float16*)alloc((size_t)768*13824*2);
  (void)ws_size; (void)in_sizes; (void)n_in; (void)out_size;

  // ---- weight conversion ----
  auto cvt = [&](const float* s, _Float16* d, int n){
    cvt_f16<<<(n+255)/256, 256, 0, stream>>>(s, d, n);
  };
  cvt(qw1, qw1h, 256*768);
  cvt(qw2, qw2h, 256*256);
  cvt(kw1, kw1h, 256*768);
  cvt(kw2, kw2h, 256*256);
  cvt(vw,  vwh,  256*768);
  cvt(ow,  owh,  768*256);
  cvt_conv_w<<<(768*13824+255)/256, 256, 0, stream>>>(bw, bwh);

  // ---- staging ----
  sim_gather<<<256, 256, 0, stream>>>(text, text_mem, memry, DhT);
  softmax_wcls<<<16384, 64, 0, stream>>>(preds, wcls);
  feats_to_xcat<<<(4*768*4096+255)/256, 256, 0, stream>>>(feats, xcat);

  // ---- sel = wcls @ D   (M=16384, N=768, K=256 padded) ----
  gemm_wmma<<<dim3(768/128, 16384/128), 256, 0, stream>>>(
      wcls, 256, DhT, 256, sel, 768, 256, 1.0f, nullptr, nullptr, 0);

  // ---- q / k / v heads ----
  gemm_wmma<<<dim3(256/128, 16384/128), 256, 0, stream>>>(
      xcat, 1536, qw1h, 768, q1b, 256, 768, 1.0f, qg1, qb1, 1);
  gemm_wmma<<<dim3(256/128, 16384/128), 256, 0, stream>>>(
      q1b, 256, qw2h, 256, qh, 256, 256, 1.0f, qg2, qb2, 1);
  gemm_wmma<<<dim3(256/128, 16384/128), 256, 0, stream>>>(
      sel, 768, kw1h, 768, k1b, 256, 768, 1.0f, kg1, kb1, 1);
  gemm_wmma<<<dim3(256/128, 16384/128), 256, 0, stream>>>(
      k1b, 256, kw2h, 256, kh, 256, 256, 1.0f, kg2, kb2, 1);
  gemm_wmma<<<dim3(256/128, 16384/128), 256, 0, stream>>>(
      sel, 768, vwh, 768, vh, 256, 768, 1.0f, vg, vb, 1);
  v_transpose<<<(4*256*4096+255)/256, 256, 0, stream>>>(vh, vT);

  // ---- attention, per batch (att tile is L2-resident: 32 MB f16) ----
  for (int b = 0; b < 4; b++){
    const _Float16* qb  = qh + (size_t)b*4096*256;
    const _Float16* kb  = kh + (size_t)b*4096*256;
    const _Float16* vb2 = vT + (size_t)b*256*4096;
    _Float16* cb = ctxh + (size_t)b*4096*256;
    // logits = (q @ k^T) * 1/16     (M=4096, N=4096, K=256)
    gemm_wmma<<<dim3(4096/128, 4096/128), 256, 0, stream>>>(
        qb, 256, kb, 256, atth, 4096, 256, 0.0625f, nullptr, nullptr, 0);
    softmax_att<<<4096, 256, 0, stream>>>(atth);
    // ctx = att @ v                 (M=4096, N=256, K=4096)
    gemm_wmma<<<dim3(256/128, 4096/128), 256, 0, stream>>>(
        atth, 4096, vb2, 4096, cb, 256, 4096, 1.0f, nullptr, nullptr, 0);
  }

  // ---- sel_out = cbr(ctx, ow) written into right half of xcat ----
  gemm_wmma<<<dim3(768/128, 16384/128), 256, 0, stream>>>(
      ctxh, 256, owh, 256, xcat + 768, 1536, 256, 1.0f, og, ob, 1);

  // ---- 3x3 conv + BN + ReLU (implicit GEMM, K = 13824) ----
  conv3x3_wmma<<<dim3(768/256, 4*64), 256, 0, stream>>>(xcat, bwh, bg, bb, out);
}
